// GPT_Layer_32444182954518
// MI455X (gfx1250) — matrix operand order repaired and checked by reading, not compiled, not run on hardware
//
#include <hip/hip_runtime.h>
#include <hip/hip_bf16.h>
#include <math.h>

// ---------------------------------------------------------------------------
// GPT layer for gfx1250 (MI455X). bf16 WMMA everywhere, fp32 accumulate,
// fp32 LN/softmax/GELU. Flash attention. TDM (tensor_load_to_lds) staging.
// ---------------------------------------------------------------------------

#define USE_TDM 1

typedef __attribute__((ext_vector_type(16))) __bf16 v16bf;
typedef __attribute__((ext_vector_type(8)))  float  v8f;
typedef __attribute__((ext_vector_type(4)))  unsigned int v4u;
typedef __attribute__((ext_vector_type(8)))  unsigned int v8u;

#define S_LEN   2048
#define D_MODEL 1024
#define N_HEADS 16
#define D_HEAD  64
#define D_FF    4096
#define N_TOK   8192          // B * S

static __device__ __forceinline__ v8f wmma_bf16(v16bf a, v16bf b, v8f c) {
    // (neg_a, A, neg_b, B, c_mod, C, reuse_a, reuse_b)
    return __builtin_amdgcn_wmma_f32_16x16x32_bf16(false, a, false, b,
                                                   (short)0, c, false, false);
}

#if USE_TDM
// Issue one TDM 2D tile load (global -> LDS). D# built per ISA 08 §8.3/8.4.
// data_size = 2 bytes. tile_d0 elements per row, tile_d1 rows, row stride in
// elements. Optional LDS padding: pad_interval v -> 2^v 8-byte units between
// pads, pad_amount v -> (v+1) DWORDs inserted.
static __device__ __forceinline__ void tdm_load_2d(
        unsigned lds_off, const void* gptr,
        unsigned tensor_d0, unsigned tensor_d1,
        unsigned tile_d0, unsigned tile_d1,
        unsigned long long stride0_elems,
        unsigned pad_en, unsigned pad_interval, unsigned pad_amount) {
    unsigned long long ga = (unsigned long long)(uintptr_t)gptr;
    v4u g0;
    g0[0] = 1u;                                             // count=1, user mode
    g0[1] = lds_off;                                        // LDS byte address
    g0[2] = (unsigned)(ga & 0xFFFFFFFFu);                   // global_addr lo
    g0[3] = (unsigned)((ga >> 32) & 0x01FFFFFFu) | 0x80000000u; // hi25 | type=2
    v8u g1;
    g1[0] = (1u << 16) | (pad_en << 20) | (pad_interval << 22) | (pad_amount << 25);
    g1[1] = (tensor_d0 & 0xFFFFu) << 16;                    // tensor_dim0[15:0]
    g1[2] = (tensor_d0 >> 16) | ((tensor_d1 & 0xFFFFu) << 16);
    g1[3] = (tensor_d1 >> 16) | ((tile_d0 & 0xFFFFu) << 16);
    g1[4] = (tile_d1 & 0xFFFFu);                            // tile_dim1, tile_dim2=0
    g1[5] = (unsigned)(stride0_elems & 0xFFFFFFFFull);      // dim0 stride lo
    g1[6] = (unsigned)((stride0_elems >> 32) & 0xFFFFull);  // dim0 stride hi
    g1[7] = 0u;
    // 2-operand form: VADDR2/VADDR3 = NULL (tensor <= 2D)
    asm volatile("tensor_load_to_lds %0, %1" :: "s"(g0), "s"(g1) : "memory");
}
#endif

// ---------------------------------------------------------------------------
// Weight conversion (fp32 -> bf16). W_q/W_k/W_v [H][D][DK] -> [D][H*DK].
// ---------------------------------------------------------------------------
__global__ void cvt_qkv_kernel(const float* __restrict__ W, __bf16* __restrict__ Wb) {
    size_t idx = (size_t)blockIdx.x * 256 + threadIdx.x;      // over D*H*DK = 1M
    int d  = (int)(idx >> 10);
    int c  = (int)(idx & 1023);
    int h  = c >> 6;
    int kk = c & 63;
    Wb[idx] = (__bf16)W[((size_t)h * D_MODEL + d) * D_HEAD + kk];
}

__global__ void cvt_plain_kernel(const float* __restrict__ W, __bf16* __restrict__ Wb, int n) {
    int idx = blockIdx.x * 256 + threadIdx.x;
    if (idx < n) Wb[idx] = (__bf16)W[idx];
}

// ---------------------------------------------------------------------------
// LayerNorm: one 256-thread block per row of 1024, fp32 in -> bf16 out.
// ---------------------------------------------------------------------------
__global__ __launch_bounds__(256)
void ln_kernel(const float* __restrict__ X, const float* __restrict__ g,
               const float* __restrict__ b, __bf16* __restrict__ Hout) {
    const int row = blockIdx.x;
    const float* x = X + (size_t)row * D_MODEL;
    float v[4];
    float s = 0.f, ss = 0.f;
#pragma unroll
    for (int i = 0; i < 4; ++i) {
        v[i] = x[threadIdx.x + i * 256];
        s  += v[i];
        ss += v[i] * v[i];
    }
#pragma unroll
    for (int off = 16; off > 0; off >>= 1) {
        s  += __shfl_xor(s,  off);
        ss += __shfl_xor(ss, off);
    }
    __shared__ float sw[8], ssw[8];
    int wid = threadIdx.x >> 5;
    if ((threadIdx.x & 31) == 0) { sw[wid] = s; ssw[wid] = ss; }
    __syncthreads();
    float ts = 0.f, tss = 0.f;
#pragma unroll
    for (int i = 0; i < 8; ++i) { ts += sw[i]; tss += ssw[i]; }
    const float mu  = ts * (1.0f / D_MODEL);
    const float var = tss * (1.0f / D_MODEL) - mu * mu;
    const float inv = rsqrtf(var + 1e-5f);
#pragma unroll
    for (int i = 0; i < 4; ++i) {
        int c = threadIdx.x + i * 256;
        float y = (v[i] - mu) * inv * g[c] + b[c];
        Hout[(size_t)row * D_MODEL + c] = (__bf16)y;
    }
}

// ---------------------------------------------------------------------------
// bf16 WMMA GEMM: C(MxN) = A(MxK) * B(KxN)  [+ epilogue]
//   block = 256 threads (8 waves), tile 128x128, BK = 32
//   wave (wm 0..3, wn 0..1) owns 32x64 = 2x4 WMMA frags (8 WMMA / K-step)
//   A tile staged by TDM (row-major + LDS padding), B tile staged transposed
//   by the threads (overlaps the in-flight TDM).
// EPI: 0 = store bf16 into [B,H,S,DK] layout (QKV)
//      1 = +bias, exact GELU, store bf16 row-major      (MLP up)
//      2 = +residual(fp32), store fp32 row-major        (O-proj)
//      3 = +bias +residual(fp32), store fp32 row-major  (MLP down)
// ---------------------------------------------------------------------------
template <int EPI>
__global__ __launch_bounds__(256)
void gemm_bf16_kernel(const __bf16* __restrict__ A, const __bf16* __restrict__ Bw,
                      const float* __restrict__ bias, const float* __restrict__ res,
                      void* __restrict__ outp, int M, int N, int K) {
    constexpr int BM = 128, BN = 128, BK = 32;
    constexpr int LDA = BK + 8;   // pad: 16 DWORDs data + 4 DWORDs pad per row
    constexpr int LDB = BK + 8;
    __shared__ __bf16 As [BM * LDA];
    __shared__ __bf16 Bst[BN * LDB];          // B tile stored transposed [n][k]

    const int tid  = threadIdx.x;
    const int lane = tid & 31;
    const int wid  = tid >> 5;
    const int wm   = wid & 3;                 // 32-row strip
    const int wn   = wid >> 2;                // 64-col strip
    const size_t rowBase = (size_t)blockIdx.y * BM;
    const size_t colBase = (size_t)blockIdx.x * BN;

    v8f acc[2][4];
#pragma unroll
    for (int mi = 0; mi < 2; ++mi)
#pragma unroll
        for (int ni = 0; ni < 4; ++ni)
#pragma unroll
            for (int i = 0; i < 8; ++i) acc[mi][ni][i] = 0.f;

    // staging assignments
    const int m_ld  = tid >> 1;               // 0..127   (manual A path)
    const int k_ld  = (tid & 1) * 16;         // 0 / 16
    const int bk_ld = tid & 31;               // 0..31
    const int bn_ld = (tid >> 5) * 16;        // 0..112 step 16

    // fragment lane decode (ISA 7.12.2)
    const int am  = lane & 15;
    const int akb = (lane < 16) ? 0 : 16;     // A: K sub-block
    const int bn  = lane & 15;
    const int bkb = (lane < 16) ? 0 : 16;     // B: K sub-block

#if USE_TDM
    const unsigned ldsA_off = (unsigned)(uintptr_t)(void*)&As[0];
#endif

    for (int k0 = 0; k0 < K; k0 += BK) {
        __syncthreads();                      // LDS reuse fence
#if USE_TDM
        if (wid == 0)                         // one wave issues the DMA
            tdm_load_2d(ldsA_off, A + rowBase * (size_t)K + k0,
                        /*tensor*/ BK, BM, /*tile*/ BK, BM,
                        (unsigned long long)K,
                        /*pad_en*/ 1u, /*interval 16DW*/ 3u, /*amount 4DW*/ 3u);
#else
        {
            const __bf16* ag = A + (rowBase + m_ld) * (size_t)K + k0 + k_ld;
            *(uint4*)&As[m_ld * LDA + k_ld]     = *(const uint4*)ag;
            *(uint4*)&As[m_ld * LDA + k_ld + 8] = *(const uint4*)(ag + 8);
        }
#endif
        // ---- stage B transposed (32x128 -> [n][k]); overlaps the TDM ----
        {
            const __bf16* bg = Bw + (size_t)(k0 + bk_ld) * N + colBase + bn_ld;
            __bf16 btmp[16];
            *(uint4*)&btmp[0] = *(const uint4*)bg;
            *(uint4*)&btmp[8] = *(const uint4*)(bg + 8);
#pragma unroll
            for (int j = 0; j < 16; ++j) Bst[(bn_ld + j) * LDB + bk_ld] = btmp[j];
            if (k0 + BK < K) __builtin_prefetch(bg + (size_t)BK * N, 0, 1);
        }
#if USE_TDM
        if (wid == 0) __builtin_amdgcn_s_wait_tensorcnt(0);
#endif
        __syncthreads();

        // ---- fragments + 8 WMMAs ----
        v16bf aF[2], bF[4];
#pragma unroll
        for (int mi = 0; mi < 2; ++mi) {
            const __bf16* ap = &As[(wm * 32 + mi * 16 + am) * LDA + akb];
#pragma unroll
            for (int i = 0; i < 8; ++i) aF[mi][i]     = ap[i];
#pragma unroll
            for (int i = 0; i < 8; ++i) aF[mi][8 + i] = ap[8 + i];
        }
#pragma unroll
        for (int ni = 0; ni < 4; ++ni) {
            const __bf16* bp = &Bst[(wn * 64 + ni * 16 + bn) * LDB + bkb];
#pragma unroll
            for (int i = 0; i < 16; ++i) bF[ni][i] = bp[i];
        }
#pragma unroll
        for (int mi = 0; mi < 2; ++mi)
#pragma unroll
            for (int ni = 0; ni < 4; ++ni)
                acc[mi][ni] = wmma_bf16(aF[mi], bF[ni], acc[mi][ni]);
    }

    // ---- epilogue ----
    const int cml   = lane & 15;
    const int cbase = (lane < 16) ? 0 : 8;
#pragma unroll
    for (int mi = 0; mi < 2; ++mi)
#pragma unroll
        for (int ni = 0; ni < 4; ++ni)
#pragma unroll
            for (int i = 0; i < 8; ++i) {
                size_t r = rowBase + wm * 32 + mi * 16 + cbase + i;
                size_t c = colBase + wn * 64 + ni * 16 + cml;
                float vacc = acc[mi][ni][i];
                if constexpr (EPI == 0) {        // QKV -> [B,H,S,DK] bf16
                    size_t b_ = r >> 11, s_ = r & 2047;
                    size_t h_ = c >> 6,  kk = c & 63;
                    ((__bf16*)outp)[(((b_ * N_HEADS + h_) * S_LEN) + s_) * D_HEAD + kk] =
                        (__bf16)vacc;
                } else if constexpr (EPI == 1) { // bias + exact GELU -> bf16
                    float xg = vacc + bias[c];
                    float y  = 0.5f * xg * (1.0f + erff(xg * 0.70710678118654752f));
                    ((__bf16*)outp)[r * (size_t)N + c] = (__bf16)y;
                } else if constexpr (EPI == 2) { // + residual -> fp32
                    ((float*)outp)[r * (size_t)N + c] = vacc + res[r * (size_t)N + c];
                } else {                          // bias + residual -> fp32
                    ((float*)outp)[r * (size_t)N + c] =
                        vacc + bias[c] + res[r * (size_t)N + c];
                }
            }
}

// ---------------------------------------------------------------------------
// Flash attention: grid (S/64, B*H), block 128 (4 waves); wave = 16 q rows.
// K/V tiles (64 keys x 64) staged via TDM; online softmax in the C-fragment
// layout; P re-enters A layout through LDS. Output bf16, head-concat layout.
// ---------------------------------------------------------------------------
__global__ __launch_bounds__(128)
void attn_kernel(const __bf16* __restrict__ Q, const __bf16* __restrict__ Kg,
                 const __bf16* __restrict__ Vg, __bf16* __restrict__ Oc) {
    const int bh   = blockIdx.y;               // b*H + h
    const int qt   = blockIdx.x;               // 64-query block
    const int tid  = threadIdx.x;
    const int lane = tid & 31;
    const int wid  = tid >> 5;
    const int qbase = qt * 64 + wid * 16;
    const size_t base = (size_t)bh * S_LEN * D_HEAD;

    __shared__ __bf16 Ks[64 * 64];
    __shared__ __bf16 Vs[64 * 64];
    __shared__ __bf16 Ps[4][16 * 64];

    const int am    = lane & 15;
    const int akb   = (lane < 16) ? 0 : 16;
    const int ncol  = lane & 15;
    const int cbase = (lane < 16) ? 0 : 8;
    const int bkb   = (lane < 16) ? 0 : 16;

    // Q fragments straight from global (contiguous 16B per group)
    v16bf aQ[2];
#pragma unroll
    for (int kk = 0; kk < 2; ++kk) {
        const __bf16* qp = Q + base + (size_t)(qbase + am) * D_HEAD + kk * 32 + akb;
#pragma unroll
        for (int i = 0; i < 8; ++i) aQ[kk][i]     = qp[i];
#pragma unroll
        for (int i = 0; i < 8; ++i) aQ[kk][8 + i] = qp[8 + i];
    }

    v8f o[4];
#pragma unroll
    for (int df = 0; df < 4; ++df)
#pragma unroll
        for (int i = 0; i < 8; ++i) o[df][i] = 0.f;
    float mrow[8], lrow[8];
#pragma unroll
    for (int i = 0; i < 8; ++i) { mrow[i] = -__builtin_inff(); lrow[i] = 0.f; }

    const float scale = 0.125f;                // 1/sqrt(64)
    const int ktiles = qt + 1;                 // causal

#if USE_TDM
    const unsigned ldsK_off = (unsigned)(uintptr_t)(void*)&Ks[0];
    const unsigned ldsV_off = (unsigned)(uintptr_t)(void*)&Vs[0];
#endif

    for (int kt = 0; kt < ktiles; ++kt) {
        const int kbase = kt * 64;
        __syncthreads();                       // K/V tile reuse fence
#if USE_TDM
        if (wid == 0) {
            tdm_load_2d(ldsK_off, Kg + base + (size_t)kbase * D_HEAD,
                        64, 64, 64, 64, 64ull, 0u, 0u, 0u);
            tdm_load_2d(ldsV_off, Vg + base + (size_t)kbase * D_HEAD,
                        64, 64, 64, 64, 64ull, 0u, 0u, 0u);
            __builtin_amdgcn_s_wait_tensorcnt(0);
        }
#else
        {
            const uint4* ksrc = (const uint4*)(Kg + base + (size_t)kbase * D_HEAD);
            const uint4* vsrc = (const uint4*)(Vg + base + (size_t)kbase * D_HEAD);
            uint4* kdst = (uint4*)Ks;
            uint4* vdst = (uint4*)Vs;
#pragma unroll
            for (int j = 0; j < 4; ++j) {
                kdst[tid + j * 128] = ksrc[tid + j * 128];
                vdst[tid + j * 128] = vsrc[tid + j * 128];
            }
        }
#endif
        __syncthreads();

        // ---- S = Q K^T (4 col-frags of 16 keys) ----
        v8f sfr[4];
#pragma unroll
        for (int nf = 0; nf < 4; ++nf) {
            v8f s;
#pragma unroll
            for (int i = 0; i < 8; ++i) s[i] = 0.f;
#pragma unroll
            for (int kk = 0; kk < 2; ++kk) {
                v16bf bK;   // B[d][key]: transposed view == Ks row-major
                const __bf16* bp = &Ks[(nf * 16 + ncol) * 64 + kk * 32 + bkb];
#pragma unroll
                for (int i = 0; i < 16; ++i) bK[i] = bp[i];
                s = wmma_bf16(aQ[kk], bK, s);
            }
            sfr[nf] = s;
        }

        // ---- scale + causal mask + tile row max ----
        const bool maskTile = (kt == ktiles - 1);
        float tmax[8];
#pragma unroll
        for (int i = 0; i < 8; ++i) tmax[i] = -__builtin_inff();
#pragma unroll
        for (int nf = 0; nf < 4; ++nf)
#pragma unroll
            for (int i = 0; i < 8; ++i) {
                float sv = sfr[nf][i] * scale;
                if (maskTile) {
                    int kglob = kbase + nf * 16 + ncol;
                    int qglob = qbase + cbase + i;
                    if (kglob > qglob) sv = -__builtin_inff();
                }
                sfr[nf][i] = sv;
                tmax[i] = fmaxf(tmax[i], sv);
            }
#pragma unroll
        for (int off = 1; off < 16; off <<= 1)
#pragma unroll
            for (int i = 0; i < 8; ++i)
                tmax[i] = fmaxf(tmax[i], __shfl_xor(tmax[i], off));

        // ---- online softmax update ----
        float corr[8], rsum[8];
#pragma unroll
        for (int i = 0; i < 8; ++i) {
            float mnew = fmaxf(mrow[i], tmax[i]);
            corr[i] = __expf(mrow[i] - mnew);
            mrow[i] = mnew;
            lrow[i] *= corr[i];
            rsum[i] = 0.f;
        }
#pragma unroll
        for (int df = 0; df < 4; ++df)
#pragma unroll
            for (int i = 0; i < 8; ++i) o[df][i] *= corr[i];

        // P = exp(S - m), stage to LDS for the layout change C -> A
#pragma unroll
        for (int nf = 0; nf < 4; ++nf)
#pragma unroll
            for (int i = 0; i < 8; ++i) {
                float p = __expf(sfr[nf][i] - mrow[i]);
                rsum[i] += p;
                Ps[wid][(cbase + i) * 64 + nf * 16 + ncol] = (__bf16)p;
            }
#pragma unroll
        for (int off = 1; off < 16; off <<= 1)
#pragma unroll
            for (int i = 0; i < 8; ++i) rsum[i] += __shfl_xor(rsum[i], off);
#pragma unroll
        for (int i = 0; i < 8; ++i) lrow[i] += rsum[i];

        // ---- O += P V ----
        v16bf aP[2];
#pragma unroll
        for (int kk = 0; kk < 2; ++kk) {
            const __bf16* pp = &Ps[wid][am * 64 + kk * 32 + akb];
#pragma unroll
            for (int i = 0; i < 8; ++i) aP[kk][i]     = pp[i];
#pragma unroll
            for (int i = 0; i < 8; ++i) aP[kk][8 + i] = pp[8 + i];
        }
#pragma unroll
        for (int df = 0; df < 4; ++df)
#pragma unroll
            for (int kk = 0; kk < 2; ++kk) {
                v16bf bV;   // B[key][d] row-major: strided gather from Vs
#pragma unroll
                for (int i = 0; i < 16; ++i)
                    bV[i] = Vs[(kk * 32 + bkb + i) * 64 + df * 16 + ncol];
                o[df] = wmma_bf16(aP[kk], bV, o[df]);
            }
    }

    // ---- normalize and write head-concat output ----
    const int b_ = bh >> 4, h_ = bh & 15;
    float inv[8];
#pragma unroll
    for (int i = 0; i < 8; ++i) inv[i] = 1.0f / lrow[i];
#pragma unroll
    for (int df = 0; df < 4; ++df)
#pragma unroll
        for (int i = 0; i < 8; ++i) {
            size_t t = (size_t)b_ * S_LEN + qbase + cbase + i;
            size_t c = (size_t)h_ * D_HEAD + df * 16 + ncol;
            Oc[t * D_MODEL + c] = (__bf16)(o[df][i] * inv[i]);
        }
}

// ---------------------------------------------------------------------------
// Host-side orchestration.
// ---------------------------------------------------------------------------
extern "C" void kernel_launch(void* const* d_in, const int* in_sizes, int n_in,
                              void* d_out, int out_size, void* d_ws, size_t ws_size,
                              hipStream_t stream) {
    const float* x    = (const float*)d_in[0];
    const float* W_q  = (const float*)d_in[1];
    const float* W_k  = (const float*)d_in[2];
    const float* W_v  = (const float*)d_in[3];
    const float* W_o  = (const float*)d_in[4];
    const float* ln1g = (const float*)d_in[5];
    const float* ln1b = (const float*)d_in[6];
    const float* ln2g = (const float*)d_in[7];
    const float* ln2b = (const float*)d_in[8];
    const float* w1   = (const float*)d_in[9];
    const float* b1   = (const float*)d_in[10];
    const float* w2   = (const float*)d_in[11];
    const float* b2   = (const float*)d_in[12];
    float* out = (float*)d_out;

    char* ws = (char*)d_ws;
    size_t off = 0;
    auto alloc = [&](size_t bytes) -> char* {
        char* p = ws + off;
        off += (bytes + 255) & ~(size_t)255;
        return p;
    };
    __bf16* h_bf  = (__bf16*)alloc((size_t)N_TOK * D_MODEL * 2);
    __bf16* wq_bf = (__bf16*)alloc((size_t)D_MODEL * D_MODEL * 2);
    __bf16* wk_bf = (__bf16*)alloc((size_t)D_MODEL * D_MODEL * 2);
    __bf16* wv_bf = (__bf16*)alloc((size_t)D_MODEL * D_MODEL * 2);
    __bf16* wo_bf = (__bf16*)alloc((size_t)D_MODEL * D_MODEL * 2);
    __bf16* w1_bf = (__bf16*)alloc((size_t)D_MODEL * D_FF * 2);
    __bf16* w2_bf = (__bf16*)alloc((size_t)D_FF * D_MODEL * 2);
    __bf16* q_bf  = (__bf16*)alloc((size_t)N_TOK * D_MODEL * 2);
    __bf16* k_bf  = (__bf16*)alloc((size_t)N_TOK * D_MODEL * 2);
    __bf16* v_bf  = (__bf16*)alloc((size_t)N_TOK * D_MODEL * 2);
    __bf16* oc_bf = (__bf16*)alloc((size_t)N_TOK * D_MODEL * 2);
    float*  x1    = (float*) alloc((size_t)N_TOK * D_MODEL * 4);
    __bf16* h2_bf = (__bf16*)alloc((size_t)N_TOK * D_MODEL * 2);
    __bf16* g_bf  = (__bf16*)alloc((size_t)N_TOK * D_FF * 2);
    if (off > ws_size) return;   // workspace too small: bail deterministically

    // 1) weights -> bf16 (QKV repacked to [D][H*DK])
    cvt_qkv_kernel<<<dim3(4096), dim3(256), 0, stream>>>(W_q, wq_bf);
    cvt_qkv_kernel<<<dim3(4096), dim3(256), 0, stream>>>(W_k, wk_bf);
    cvt_qkv_kernel<<<dim3(4096), dim3(256), 0, stream>>>(W_v, wv_bf);
    cvt_plain_kernel<<<dim3(4096),  dim3(256), 0, stream>>>(W_o, wo_bf, D_MODEL * D_MODEL);
    cvt_plain_kernel<<<dim3(16384), dim3(256), 0, stream>>>(w1, w1_bf, D_MODEL * D_FF);
    cvt_plain_kernel<<<dim3(16384), dim3(256), 0, stream>>>(w2, w2_bf, D_FF * D_MODEL);

    // 2) LN1
    ln_kernel<<<dim3(N_TOK), dim3(256), 0, stream>>>(x, ln1g, ln1b, h_bf);

    // 3) Q/K/V projections: [8192x1024] x [1024x1024], out in [B,H,S,DK]
    dim3 gproj(D_MODEL / 128, N_TOK / 128);
    gemm_bf16_kernel<0><<<gproj, dim3(256), 0, stream>>>(h_bf, wq_bf, nullptr, nullptr,
                                                         q_bf, N_TOK, D_MODEL, D_MODEL);
    gemm_bf16_kernel<0><<<gproj, dim3(256), 0, stream>>>(h_bf, wk_bf, nullptr, nullptr,
                                                         k_bf, N_TOK, D_MODEL, D_MODEL);
    gemm_bf16_kernel<0><<<gproj, dim3(256), 0, stream>>>(h_bf, wv_bf, nullptr, nullptr,
                                                         v_bf, N_TOK, D_MODEL, D_MODEL);

    // 4) causal flash attention
    attn_kernel<<<dim3(S_LEN / 64, 4 * N_HEADS), dim3(128), 0, stream>>>(q_bf, k_bf, v_bf, oc_bf);

    // 5) O projection + residual -> x1 (fp32)
    gemm_bf16_kernel<2><<<gproj, dim3(256), 0, stream>>>(oc_bf, wo_bf, nullptr, x,
                                                         x1, N_TOK, D_MODEL, D_MODEL);

    // 6) LN2
    ln_kernel<<<dim3(N_TOK), dim3(256), 0, stream>>>(x1, ln2g, ln2b, h2_bf);

    // 7) MLP up + bias + exact GELU -> bf16
    dim3 gmlp1(D_FF / 128, N_TOK / 128);
    gemm_bf16_kernel<1><<<gmlp1, dim3(256), 0, stream>>>(h2_bf, w1_bf, b1, nullptr,
                                                         g_bf, N_TOK, D_FF, D_MODEL);

    // 8) MLP down + bias + residual -> d_out (fp32)
    gemm_bf16_kernel<3><<<gproj, dim3(256), 0, stream>>>(g_bf, w2_bf, b2, x1,
                                                         out, N_TOK, D_MODEL, D_FF);
}
